// GATv2_15530601742538
// MI455X (gfx1250) — matrix operand order repaired and checked by reading, not compiled, not run on hardware
//
#include <hip/hip_runtime.h>

typedef __attribute__((ext_vector_type(16))) __bf16        v16bf;
typedef __attribute__((ext_vector_type(8)))  float         v8f;
typedef __attribute__((ext_vector_type(8)))  unsigned int  v8u;

#define HC  256   // H*C
#define H_  4
#define C_  64
#define NG_ 256

// ---------- helpers ----------
__device__ __forceinline__ unsigned f2bf(float f) {
  unsigned u = __float_as_uint(f);
  return ((u + 0x7FFFu + ((u >> 16) & 1u)) >> 16) & 0xFFFFu;   // RNE fp32->bf16
}
// monotonic float<->uint encoding so atomicMax(uint) == float max
__device__ __forceinline__ unsigned fenc(float f) {
  unsigned u = __float_as_uint(f);
  return (u & 0x80000000u) ? ~u : (u | 0x80000000u);
}
__device__ __forceinline__ float fdec(unsigned u) {
  return __uint_as_float((u & 0x80000000u) ? (u & 0x7FFFFFFFu) : ~u);
}
#define ENC_NEG_INF 0x007FFFFFu   // fenc(-inf)

__device__ __forceinline__ v8u pack8(uint4 a, uint4 b) {
  v8u r;
  r[0] = a.x; r[1] = a.y; r[2] = a.z; r[3] = a.w;
  r[4] = b.x; r[5] = b.y; r[6] = b.z; r[7] = b.w;
  return r;
}

// ---------- weight pack: fp32 [K,256] -> bf16 B-fragment layout ----------
// layout: [ktile][ntile][lane(32)][v(8)] dwords; lane<16 holds K=k0..k0+15,
// lane>=16 holds K=k0+16..k0+31 (2 bf16 per dword, consecutive K).
__global__ void pack_w_kernel(const float* __restrict__ W, unsigned* __restrict__ P, int K) {
  int idx = blockIdx.x * 256 + threadIdx.x;
  int total = (K >> 5) << 12;             // (K/32)*16*32*8
  if (idx >= total) return;
  int v     = idx & 7;
  int lane  = (idx >> 3) & 31;
  int ntile = (idx >> 8) & 15;
  int ktile = idx >> 12;
  int khalf = lane >> 4;
  int n = ntile * 16 + (lane & 15);
  int k = ktile * 32 + khalf * 16 + 2 * v;
  unsigned lo = f2bf(W[(size_t)k * HC + n]);
  unsigned hi = f2bf(W[(size_t)(k + 1) * HC + n]);
  P[idx] = lo | (hi << 16);
}

// ---------- fp32 -> bf16 activation convert (zero-fills M padding) ----------
__global__ void cvt_bf16_kernel(const float* __restrict__ in, unsigned short* __restrict__ out,
                                int ntotal, int nvalid) {
  int i = blockIdx.x * 256 + threadIdx.x;
  if (i < ntotal) out[i] = (i < nvalid) ? (unsigned short)f2bf(in[i]) : (unsigned short)0;
}

// ---------- WMMA GEMM: C[Mpad,256] = bf16(A[Mpad,K]) @ bf16(W[K,256]) + bias ----------
// M is padded to a multiple of 16 -> fully branchless: no predication, no tails.
// One wave computes a 16x64 strip (4 independent 16x16 WMMA tiles sharing the
// A fragment) -> no WMMA->WMMA accumulator hazards, 4x A reuse.
__global__ void __launch_bounds__(256) gemm_wmma_kernel(
    const unsigned short* __restrict__ Abf, const unsigned* __restrict__ Bpack,
    const float* __restrict__ bias, float* __restrict__ Cout, int Mpad, int K) {
  int lane  = threadIdx.x & 31;
  int wave  = threadIdx.x >> 5;
  int t = blockIdx.x * 8 + wave;
  if (t >= (Mpad >> 4) * 4) return;        // wave-uniform
  int mtile = t >> 2;
  int ntb   = (t & 3) << 2;                // ntile base: 0,4,8,12
  int khalf = lane >> 4;
  int m     = lane & 15;
  int r     = mtile * 16 + m;
  const uint4* Arow  = (const uint4*)(Abf + (size_t)r * K);
  const uint4* Bbase = (const uint4*)Bpack;

  v8f acc[4] = {};
  for (int k0 = 0; k0 < K; k0 += 32) {
    uint4 a0 = Arow[(k0 >> 3) + 2 * khalf];        // K = k0 + khalf*16 + {0..7}
    uint4 a1 = Arow[(k0 >> 3) + 2 * khalf + 1];    // K = k0 + khalf*16 + 8 + {0..7}
    v16bf av = __builtin_bit_cast(v16bf, pack8(a0, a1));
    // uint4 index: ((ktile*16 + ntile)*32 + lane) * 2 ; +64 per ntile
    const uint4* bp = Bbase + (((size_t)(k0 >> 5) * 16 + ntb) * 32 + lane) * 2;
#pragma unroll
    for (int j = 0; j < 4; ++j) {
      uint4 b0 = bp[j * 64 + 0];
      uint4 b1 = bp[j * 64 + 1];
      acc[j] = __builtin_amdgcn_wmma_f32_16x16x32_bf16(
          false, av,
          false, __builtin_bit_cast(v16bf, pack8(b0, b1)),
          (short)0, acc[j], false, false);
    }
  }

  int colb = ntb * 16 + m;
#pragma unroll
  for (int j = 0; j < 4; ++j) {
    int col = colb + j * 16;
    float bv = bias[col];
#pragma unroll
    for (int v = 0; v < 8; ++v) {
      int rr = mtile * 16 + v + (khalf << 3);
      Cout[(size_t)rr * HC + col] = acc[j][v] + bv;
    }
  }
}

// ---------- per-layer scratch init (zeroes padded accumulator rows too) ----------
__global__ void init_layer_kernel(float* __restrict__ yacc, unsigned* __restrict__ lmax,
                                  float* __restrict__ den, int ny, int nh) {
  int i = blockIdx.x * 256 + threadIdx.x;
  if (i < ny) yacc[i] = 0.f;
  if (i < nh) { lmax[i] = ENC_NEG_INF; den[i] = 0.f; }
}

// ---------- edge phase 1: logits + segment max (wave per (edge,head)) ----------
__global__ void edge_logits_kernel(const float* __restrict__ xl, const float* __restrict__ xr,
                                   const float* __restrict__ att, const int* __restrict__ ei,
                                   int E, int Etot, float* __restrict__ logits,
                                   unsigned* __restrict__ lmax) {
  int wave = blockIdx.x * 8 + ((int)threadIdx.x >> 5);
  int lane = threadIdx.x & 31;
  if (wave >= Etot * H_) return;
  int e = wave >> 2;                       // H_ == 4
  int h = wave & 3;
  int s, d;
  if (e < E) { s = ei[e]; d = ei[E + e]; } else { s = e - E; d = s; }
  int base = h * C_;
  float m0 = xl[(size_t)s * HC + base + lane]      + xr[(size_t)d * HC + base + lane];
  float m1 = xl[(size_t)s * HC + base + lane + 32] + xr[(size_t)d * HC + base + lane + 32];
  m0 = (m0 > 0.f) ? m0 : 0.2f * m0;        // leaky_relu
  m1 = (m1 > 0.f) ? m1 : 0.2f * m1;
  float acc = m0 * att[base + lane] + m1 * att[base + lane + 32];
#pragma unroll
  for (int off = 16; off > 0; off >>= 1) acc += __shfl_xor(acc, off);
  if (lane == 0) {
    logits[(size_t)e * H_ + h] = acc;
    atomicMax(&lmax[(size_t)d * H_ + h], fenc(acc));
  }
}

// ---------- edge phase 2: exp + denominator ----------
__global__ void edge_exp_kernel(const int* __restrict__ ei, int E, int Etot,
                                float* __restrict__ exbuf, const unsigned* __restrict__ lmax,
                                float* __restrict__ den) {
  int idx = blockIdx.x * 256 + threadIdx.x;
  if (idx >= Etot * H_) return;
  int e = idx >> 2, h = idx & 3;
  int d = (e < E) ? ei[E + e] : (e - E);
  float ex = __expf(exbuf[idx] - fdec(lmax[(size_t)d * H_ + h]));
  exbuf[idx] = ex;
  atomicAdd(&den[(size_t)d * H_ + h], ex);
}

// ---------- edge phase 3: weighted aggregation (thread per edge-channel-quad) ----------
__global__ void edge_agg_kernel(const int* __restrict__ ei, int E, int Etot,
                                const float* __restrict__ xl, const float* __restrict__ exbuf,
                                const float* __restrict__ den, float* __restrict__ yacc) {
  int idx = blockIdx.x * 256 + threadIdx.x;
  if (idx >= Etot * (HC / 4)) return;
  int e = idx >> 6;                        // HC/4 == 64 quads per edge
  int q = idx & 63;
  int c = q << 2;
  int h = c >> 6;
  int s, d;
  if (e < E) { s = ei[e]; d = ei[E + e]; } else { s = e - E; d = s; }
  float alpha = exbuf[(size_t)e * H_ + h] / fmaxf(den[(size_t)d * H_ + h], 1e-16f);
  float4 xv = *(const float4*)(xl + (size_t)s * HC + c);
  float* yp = yacc + (size_t)d * HC + c;
  atomicAdd(yp + 0, alpha * xv.x);
  atomicAdd(yp + 1, alpha * xv.y);
  atomicAdd(yp + 2, alpha * xv.z);
  atomicAdd(yp + 3, alpha * xv.w);
}

__global__ void add_bias_kernel(float* __restrict__ y, const float* __restrict__ bias, int n) {
  int i = blockIdx.x * 256 + threadIdx.x;
  if (i < n) y[i] += bias[i & (HC - 1)];
}

__global__ void mean_bias_kernel(const float* __restrict__ yacc, const float* __restrict__ bias,
                                 float* __restrict__ y2, int N) {
  int i = blockIdx.x * 256 + threadIdx.x;
  if (i >= N * C_) return;
  int n = i >> 6, c = i & 63;
  float s = 0.f;
#pragma unroll
  for (int h = 0; h < H_; ++h) s += yacc[(size_t)n * HC + h * C_ + c];
  y2[i] = 0.25f * s + bias[c];
}

// ---------- readout ----------
__global__ void init_readout_kernel(unsigned* __restrict__ o1, float* __restrict__ o2, int n) {
  int i = blockIdx.x * 256 + threadIdx.x;
  if (i < n) { o1[i] = ENC_NEG_INF; o2[i] = 0.f; }
}

__global__ void readout_nodes_kernel(const float* __restrict__ y2, const int* __restrict__ batch,
                                     const int* __restrict__ comp, const float* __restrict__ ww,
                                     const float* __restrict__ wb, unsigned* __restrict__ out1enc,
                                     float* __restrict__ out2, int N) {
  int n = blockIdx.x * 8 + ((int)threadIdx.x >> 5);
  int lane = threadIdx.x & 31;
  if (n >= N) return;
  float x0 = y2[(size_t)n * C_ + lane];
  float x1 = y2[(size_t)n * C_ + lane + 32];
  float acc = x0 * ww[lane] + x1 * ww[lane + 32];
#pragma unroll
  for (int off = 16; off > 0; off >>= 1) acc += __shfl_xor(acc, off);
  float w = 1.f / (1.f + __expf(-(acc + wb[0])));       // sigmoid gate, all lanes
  int seg = batch[n] + comp[n] * NG_;
  atomicMax(&out1enc[(size_t)seg * C_ + lane],      fenc(x0));
  atomicMax(&out1enc[(size_t)seg * C_ + lane + 32], fenc(x1));
  atomicAdd(&out2[(size_t)seg * C_ + lane],      w * x0);
  atomicAdd(&out2[(size_t)seg * C_ + lane + 32], w * x1);
}

// ---------- final MLP: block per graph ----------
__global__ void mlp_kernel(const unsigned* __restrict__ out1enc, const float* __restrict__ out2,
                           const float* __restrict__ m1w, const float* __restrict__ m1b,
                           const float* __restrict__ prelu_a, const float* __restrict__ m2w,
                           const float* __restrict__ m2b, float* __restrict__ dout) {
  __shared__ float feat[4 * C_];
  __shared__ float red[128];
  int g = blockIdx.x;
  int t = threadIdx.x;
  for (int j = t; j < 4 * C_; j += 128) {
    int half = j >> 7;                         // 0: graphs [0,ng), 1: [ng,2ng)
    int jj = j & 127;
    int segg = g + half * NG_;
    feat[j] = (jj < C_) ? fdec(out1enc[(size_t)segg * C_ + jj])
                        : out2[(size_t)segg * C_ + (jj - C_)];
  }
  __syncthreads();
  float a = *prelu_a;
  float h = m1b[t];
  for (int j = 0; j < 4 * C_; ++j) h += feat[j] * m1w[j * 128 + t];
  h = (h >= 0.f) ? h : a * h;                  // PReLU
  red[t] = h * m2w[t];
  __syncthreads();
  for (int s = 64; s > 0; s >>= 1) { if (t < s) red[t] += red[t + s]; __syncthreads(); }
  if (t == 0) dout[g] = red[0] + m2b[0];
}

// ---------- driver ----------
extern "C" void kernel_launch(void* const* d_in, const int* in_sizes, int n_in,
                              void* d_out, int out_size, void* d_ws, size_t ws_size,
                              hipStream_t stream) {
  const float* x      = (const float*)d_in[0];
  const int*   ei     = (const int*)d_in[1];
  const int*   batch  = (const int*)d_in[2];
  const int*   comp   = (const int*)d_in[3];
  const float* Wl[3]  = {(const float*)d_in[5],  (const float*)d_in[11], (const float*)d_in[17]};
  const float* bl[3]  = {(const float*)d_in[6],  (const float*)d_in[12], (const float*)d_in[18]};
  const float* Wr[3]  = {(const float*)d_in[7],  (const float*)d_in[13], (const float*)d_in[19]};
  const float* br[3]  = {(const float*)d_in[8],  (const float*)d_in[14], (const float*)d_in[20]};
  const float* att[3] = {(const float*)d_in[9],  (const float*)d_in[15], (const float*)d_in[21]};
  const float* bia[3] = {(const float*)d_in[10], (const float*)d_in[16], (const float*)d_in[22]};
  const float* ww  = (const float*)d_in[23];
  const float* wb  = (const float*)d_in[24];
  const float* m1w = (const float*)d_in[25];
  const float* m1b = (const float*)d_in[26];
  const float* pra = (const float*)d_in[27];
  const float* m2w = (const float*)d_in[28];
  const float* m2b = (const float*)d_in[29];
  (void)n_in; (void)out_size; (void)ws_size;

  int N = in_sizes[0] / 128;
  int E = in_sizes[1] / 2;
  int Etot = E + N;
  int Mpad = (N + 15) & ~15;               // pad rows so GEMM is branchless

  char* p = (char*)d_ws;
  auto take = [&](size_t bytes) -> void* {
    void* r = (void*)p;
    p += (bytes + 255) & ~(size_t)255;
    return r;
  };
  unsigned short* xbf  = (unsigned short*)take((size_t)Mpad * HC * 2);
  float*    xl    = (float*)take((size_t)Mpad * HC * 4);
  float*    xr    = (float*)take((size_t)Mpad * HC * 4);
  float*    yacc  = (float*)take((size_t)Mpad * HC * 4);
  float*    y2    = (float*)take((size_t)N * C_ * 4);
  float*    exbuf = (float*)take((size_t)Etot * H_ * 4);
  unsigned* lmax  = (unsigned*)take((size_t)N * H_ * 4);
  float*    den   = (float*)take((size_t)N * H_ * 4);
  int Ks[3] = {128, HC, HC};
  unsigned* pw[6];
  for (int i = 0; i < 3; ++i) {
    pw[2 * i]     = (unsigned*)take((size_t)Ks[i] * HC * 2);
    pw[2 * i + 1] = (unsigned*)take((size_t)Ks[i] * HC * 2);
  }
  unsigned* out1enc = (unsigned*)take((size_t)2 * NG_ * C_ * 4);
  float*    out2    = (float*)take((size_t)2 * NG_ * C_ * 4);

  // pack all weights to bf16 fragment layout
  for (int i = 0; i < 3; ++i) {
    int tot = Ks[i] * HC / 2;
    pack_w_kernel<<<(tot + 255) / 256, 256, 0, stream>>>(Wl[i], pw[2 * i], Ks[i]);
    pack_w_kernel<<<(tot + 255) / 256, 256, 0, stream>>>(Wr[i], pw[2 * i + 1], Ks[i]);
  }

  const float* layer_in = x;
  int in_rows = N;                         // valid rows in layer_in
  for (int i = 0; i < 3; ++i) {
    int K = Ks[i];
    int ntotal = Mpad * K;
    int nvalid = in_rows * K;
    cvt_bf16_kernel<<<(ntotal + 255) / 256, 256, 0, stream>>>(layer_in, xbf, ntotal, nvalid);

    int gwaves = (Mpad / 16) * 4;
    gemm_wmma_kernel<<<(gwaves + 7) / 8, 256, 0, stream>>>(xbf, pw[2 * i],     bl[i], xl, Mpad, K);
    gemm_wmma_kernel<<<(gwaves + 7) / 8, 256, 0, stream>>>(xbf, pw[2 * i + 1], br[i], xr, Mpad, K);

    int ny = Mpad * HC;                    // zero pad rows too (kept zero for next cvt)
    init_layer_kernel<<<(ny + 255) / 256, 256, 0, stream>>>(yacc, lmax, den, ny, N * H_);

    int totw = Etot * H_;
    edge_logits_kernel<<<(totw + 7) / 8, 256, 0, stream>>>(xl, xr, att[i], ei, E, Etot, exbuf, lmax);
    edge_exp_kernel<<<(totw + 255) / 256, 256, 0, stream>>>(ei, E, Etot, exbuf, lmax, den);
    int tota = Etot * (HC / 4);
    edge_agg_kernel<<<(tota + 255) / 256, 256, 0, stream>>>(ei, E, Etot, xl, exbuf, den, yacc);

    if (i < 2) {
      add_bias_kernel<<<(N * HC + 255) / 256, 256, 0, stream>>>(yacc, bia[i], N * HC);
      layer_in = yacc;
      in_rows = Mpad;                      // pad rows of yacc are valid zeros
    } else {
      mean_bias_kernel<<<(N * C_ + 255) / 256, 256, 0, stream>>>(yacc, bia[i], y2, N);
    }
  }

  init_readout_kernel<<<(2 * NG_ * C_ + 255) / 256, 256, 0, stream>>>(out1enc, out2, 2 * NG_ * C_);
  readout_nodes_kernel<<<(N + 7) / 8, 256, 0, stream>>>(y2, batch, comp, ww, wb, out1enc, out2, N);
  mlp_kernel<<<NG_, 128, 0, stream>>>(out1enc, out2, m1w, m1b, pra, m2w, m2b, (float*)d_out);
}
